// Model_18442589569624
// MI455X (gfx1250) — matrix operand order repaired and checked by reading, not compiled, not run on hardware
//
#include <hip/hip_runtime.h>

// ---------------------------------------------------------------------------
// RAFT forward for MI455X (gfx1250). All convolutions + the correlation
// volume run through one f16 WMMA GEMM (v_wmma_f32_16x16x32_f16, f32 acc).
// GEMM computes C = A * B^T with A:[M][Kp], B:[N][Kp] f16 row-major, Kp
// padded to 32 so the K dimension needs no bounds checks. Interior tiles are
// staged to LDS with GLOBAL_LOAD_ASYNC_TO_LDS_B128 (ASYNCcnt) when the
// builtin exists, else aligned b128 load/store. Fragments are read from LDS
// as contiguous ds_load_b128 pairs per the ISA 7.12.2 wave32 layouts.
// iters hard-coded to 12 (reading the device scalar would break capture).
// ---------------------------------------------------------------------------

typedef _Float16 half_t;
typedef __attribute__((ext_vector_type(16))) _Float16 v16h;
typedef __attribute__((ext_vector_type(8)))  _Float16 v8h;
typedef __attribute__((ext_vector_type(8)))  float    v8f;
typedef __attribute__((ext_vector_type(4)))  int      v4i;

#define NB   2
#define H1   60
#define W1   80
#define HW1  (H1 * W1)
#define ITERS 12

#if defined(__gfx1250__) && __has_builtin(__builtin_amdgcn_global_load_async_to_lds_b128) && __has_builtin(__builtin_amdgcn_s_wait_asynccnt)
#define HAVE_ASYNC_LDS 1
#define LDSV(p) ((__attribute__((address_space(3))) v4i*)(void*)(p))
#endif

static inline unsigned cdivu(long long a, long long b) {
  return (unsigned)((a + b - 1) / b);
}
static inline int kpad(int k) { return (k + 31) & ~31; }

// ------------------------------ WMMA GEMM ----------------------------------
// C[M,N] = act(scale * (sum_k A[m,k]*B[n,k] + bias[n]))
// Block = 256 threads = 8 waves (2x4 wave tiles of 16x16) -> 32x64 per block.
// K-step = 64 (two WMMAs per wave per barrier).
// pix > 0 => NCHW store: m = n_img*pix + p, out ((n_img*Cout + col)*pix + p).
__global__ __launch_bounds__(256)
void k_wmma_gemm(const half_t* __restrict__ A, const half_t* __restrict__ Bm,
                 const float* __restrict__ bias, float* __restrict__ C,
                 int M, int N, int Kp,
                 long long sA, long long sB, long long sC,
                 float scale, int act, int pix, int Cout)
{
  A  += (long long)blockIdx.z * sA;
  Bm += (long long)blockIdx.z * sB;
  C  += (long long)blockIdx.z * sC;

  __shared__ alignas(16) half_t Asp[32][72];  // [m][k], 8-half pad vs banks
  __shared__ alignas(16) half_t Bsp[64][72];  // [n][k]

  const int bm   = blockIdx.y * 32;
  const int bn   = blockIdx.x * 64;
  const int lane = threadIdx.x & 31;
  const int wave = threadIdx.x >> 5;
  const int wm   = wave >> 2;            // 0..1
  const int wn   = wave & 3;             // 0..3
  const int l15  = lane & 15;
  const int hi   = lane >> 4;            // 0 or 1
  const int mrow = wm * 16 + l15;
  const int ncol = wn * 16 + l15;
  const bool interior = (bm + 32 <= M) && (bn + 64 <= N);

  // cooperative-fill coordinates (per thread, 128-bit granules)
  const int ar = threadIdx.x >> 3;          // 0..31   A row
  const int ak = (threadIdx.x & 7) * 8;     // 0..56   A k offset (8 halves)
  const int br = threadIdx.x >> 2;          // 0..63   B row
  const int bk = (threadIdx.x & 3) * 16;    // 0,16,32,48 (16 halves)

  const v8h VZ = {(half_t)0, (half_t)0, (half_t)0, (half_t)0,
                  (half_t)0, (half_t)0, (half_t)0, (half_t)0};

  v8f acc = {0.f, 0.f, 0.f, 0.f, 0.f, 0.f, 0.f, 0.f};

  for (int k0 = 0; k0 < Kp; k0 += 64) {
    const half_t* ga = A  + (long long)(bm + ar) * Kp + k0 + ak;
    const half_t* gb = Bm + (long long)(bn + br) * Kp + k0 + bk;
    if (interior) {
#ifdef HAVE_ASYNC_LDS
      __builtin_amdgcn_global_load_async_to_lds_b128(
          (v4i*)ga, LDSV(&Asp[ar][ak]), 0, 0);
      __builtin_amdgcn_global_load_async_to_lds_b128(
          (v4i*)gb, LDSV(&Bsp[br][bk]), 0, 0);
      __builtin_amdgcn_global_load_async_to_lds_b128(
          (v4i*)(gb + 8), LDSV(&Bsp[br][bk + 8]), 0, 0);
      if (k0 + 64 < Kp) __builtin_prefetch(ga + 64, 0, 0);
      __builtin_amdgcn_s_wait_asynccnt(0);
#else
      *(v8h*)&Asp[ar][ak]     = *(const v8h*)ga;
      *(v8h*)&Bsp[br][bk]     = *(const v8h*)gb;
      *(v8h*)&Bsp[br][bk + 8] = *(const v8h*)(gb + 8);
      if (k0 + 64 < Kp) __builtin_prefetch(ga + 64, 0, 0);
#endif
    } else {
      *(v8h*)&Asp[ar][ak]     = (bm + ar < M) ? *(const v8h*)ga       : VZ;
      *(v8h*)&Bsp[br][bk]     = (bn + br < N) ? *(const v8h*)gb       : VZ;
      *(v8h*)&Bsp[br][bk + 8] = (bn + br < N) ? *(const v8h*)(gb + 8) : VZ;
    }
    __syncthreads();

    // Fragments per ISA 7.12.2 (wave32):
    //  A 16x32: lanes 0-15 hold K=0..7 & 16..23 of row m; lanes 16-31 hold
    //           K=8..15 & 24..31.  -> two contiguous b128 reads.
    //  B 32x16 (as B^T rows): lanes 0-15 hold K=0..7 & 16..23 of row n; lanes
    //           16-31 hold K=8..15 & 24..31. -> two contiguous b128 reads.
    #pragma unroll
    for (int c = 0; c < 2; ++c) {
      const int kc = c * 32;
      v8h alo = *(const v8h*)&Asp[mrow][kc + hi * 8];
      v8h ahi = *(const v8h*)&Asp[mrow][kc + hi * 8 + 16];
      v8h blo = *(const v8h*)&Bsp[ncol][kc + hi * 8];
      v8h bhi = *(const v8h*)&Bsp[ncol][kc + hi * 8 + 16];
      v16h a = __builtin_shufflevector(alo, ahi, 0,1,2,3,4,5,6,7,8,9,10,11,12,13,14,15);
      v16h b = __builtin_shufflevector(blo, bhi, 0,1,2,3,4,5,6,7,8,9,10,11,12,13,14,15);
      acc = __builtin_amdgcn_wmma_f32_16x16x32_f16(
          false, a, false, b, (short)0, acc, false, false);
    }
    __syncthreads();
  }

  // Epilogue. C/D layout: VGPR r -> M = r + 8*hi, N = lane&15.
  const int gn = bn + ncol;
  if (gn >= N) return;
  const float bv = bias ? bias[gn] : 0.f;
  #pragma unroll
  for (int r = 0; r < 8; ++r) {
    int gm = bm + wm * 16 + hi * 8 + r;
    if (gm >= M) continue;
    float v = scale * (acc[r] + bv);
    if      (act == 1) v = fmaxf(v, 0.f);
    else if (act == 2) v = tanhf(v);
    else if (act == 3) v = 1.f / (1.f + __expf(-v));
    long long o;
    if (pix > 0) {
      int nb = gm / pix, pp = gm % pix;
      o = ((long long)nb * Cout + gn) * pix + pp;
    } else {
      o = (long long)gm * N + gn;
    }
    C[o] = v;
  }
}

// ----------------------------- data movers ---------------------------------
// src f32 [R][C] -> dst f16 [C][R]   (fmap [256][HW] -> [HW][256])
__global__ void k_tconv_f16(const float* __restrict__ s, half_t* __restrict__ d,
                            int R, int C) {
  long long i = blockIdx.x * 256LL + threadIdx.x;
  long long tot = (long long)R * C;
  if (i >= tot) return;
  int r = (int)(i / C), c = (int)(i % C);
  d[(long long)c * R + r] = (half_t)s[i];
}

// weights [Cout][Kd] f32 -> [Cout][Kp] f16, zero padded
__global__ void k_wpad_f16(const float* __restrict__ w, half_t* __restrict__ o,
                           int Cout, int Kd, int Kp) {
  long long i = blockIdx.x * 256LL + threadIdx.x;
  long long tot = (long long)Cout * Kp;
  if (i >= tot) return;
  int kk = (int)(i % Kp);
  int co = (int)(i / Kp);
  o[i] = (kk < Kd) ? (half_t)w[(long long)co * Kd + kk] : (half_t)0.f;
}

__global__ void k_im2col_f16(const float* __restrict__ x, half_t* __restrict__ col,
                             int Nb, int Cin, int Hin, int Win, int OH, int OW,
                             int kh, int kw, int stride, int ph, int pw,
                             int Kd, int Kp)
{
  long long M = (long long)Nb * OH * OW;
  long long tot = M * Kp;
  long long idx = blockIdx.x * 256LL + threadIdx.x;
  if (idx >= tot) return;
  long long m = idx / Kp;
  int k = (int)(idx % Kp);
  float v = 0.f;
  if (k < Kd) {
    int ox = (int)(m % OW);
    int oy = (int)((m / OW) % OH);
    int n  = (int)(m / ((long long)OW * OH));
    int kx = k % kw, ky = (k / kw) % kh, c = k / (kw * kh);
    int ix = ox * stride + kx - pw;
    int iy = oy * stride + ky - ph;
    if (ix >= 0 && ix < Win && iy >= 0 && iy < Hin)
      v = x[((long long)(n * Cin + c) * Hin + iy) * Win + ix];
  }
  col[idx] = (half_t)v;
}

// ----------------------------- norm / pool ---------------------------------
__global__ __launch_bounds__(256)
void k_instnorm_relu(float* __restrict__ x, int HW) {
  float* p = x + (long long)blockIdx.x * HW;
  __shared__ float ss[256], sq[256];
  float a = 0.f, b = 0.f;
  for (int i = threadIdx.x; i < HW; i += 256) { float v = p[i]; a += v; b += v * v; }
  ss[threadIdx.x] = a; sq[threadIdx.x] = b;
  __syncthreads();
  for (int st = 128; st > 0; st >>= 1) {
    if ((int)threadIdx.x < st) { ss[threadIdx.x] += ss[threadIdx.x + st];
                                 sq[threadIdx.x] += sq[threadIdx.x + st]; }
    __syncthreads();
  }
  float mu  = ss[0] / HW;
  float var = sq[0] / HW - mu * mu;
  float rs  = rsqrtf(var + 1e-5f);
  for (int i = threadIdx.x; i < HW; i += 256)
    p[i] = fmaxf((p[i] - mu) * rs, 0.f);
}

__global__ void k_avgpool2(const float* __restrict__ in, float* __restrict__ o,
                           long long NC, int h, int w) {
  int ho = h / 2, wo = w / 2;
  long long tot = NC * ho * wo;
  long long idx = blockIdx.x * 256LL + threadIdx.x;
  if (idx >= tot) return;
  long long n = idx / ((long long)ho * wo);
  int rem = (int)(idx % ((long long)ho * wo));
  int y = rem / wo, x = rem % wo;
  const float* p = in + n * (long long)h * w;
  o[idx] = 0.25f * (p[(2*y)*w + 2*x] + p[(2*y)*w + 2*x + 1] +
                    p[(2*y+1)*w + 2*x] + p[(2*y+1)*w + 2*x + 1]);
}

// ----------------------------- flow / coords --------------------------------
__global__ void k_init_coords(float* __restrict__ c) {
  int idx = blockIdx.x * 256 + threadIdx.x;
  if (idx >= NB * 2 * HW1) return;
  int x = idx % W1, y = (idx / W1) % H1, ch = (idx / HW1) % 2;
  c[idx] = ch == 0 ? (float)x : (float)y;
}

__global__ void k_sub_grid(const float* __restrict__ c, float* __restrict__ f) {
  int idx = blockIdx.x * 256 + threadIdx.x;
  if (idx >= NB * 2 * HW1) return;
  int x = idx % W1, y = (idx / W1) % H1, ch = (idx / HW1) % 2;
  f[idx] = c[idx] - (ch ? (float)y : (float)x);
}

__global__ void k_ewise_add(float* __restrict__ a, const float* __restrict__ b,
                            long long n) {
  long long i = blockIdx.x * 256LL + threadIdx.x;
  if (i < n) a[i] += b[i];
}

__global__ void k_ewise_mul(const float* __restrict__ a, const float* __restrict__ b,
                            float* __restrict__ o, long long n) {
  long long i = blockIdx.x * 256LL + threadIdx.x;
  if (i < n) o[i] = a[i] * b[i];
}

__global__ void k_gru_combine(float* __restrict__ h, const float* __restrict__ z,
                              const float* __restrict__ q, long long n) {
  long long i = blockIdx.x * 256LL + threadIdx.x;
  if (i < n) h[i] = (1.f - z[i]) * h[i] + z[i] * q[i];
}

__global__ void k_split_cnet(const float* __restrict__ cnet, float* __restrict__ net,
                             float* __restrict__ inp) {
  long long i = blockIdx.x * 256LL + threadIdx.x;
  long long tot = (long long)NB * 128 * HW1;
  if (i >= tot) return;
  int p = (int)(i % HW1);
  int c = (int)((i / HW1) % 128);
  int b = (int)(i / ((long long)128 * HW1));
  net[i] = tanhf(cnet[((long long)b * 256 + c) * HW1 + p]);
  inp[i] = fmaxf(cnet[((long long)b * 256 + 128 + c) * HW1 + p], 0.f);
}

__global__ void k_concat2(const float* __restrict__ a, int Ca,
                          const float* __restrict__ b, int Cb,
                          float* __restrict__ o, int hw) {
  long long tot = (long long)NB * (Ca + Cb) * hw;
  long long i = blockIdx.x * 256LL + threadIdx.x;
  if (i >= tot) return;
  int p = (int)(i % hw);
  long long r = i / hw;
  int c = (int)(r % (Ca + Cb));
  int n = (int)(r / (Ca + Cb));
  o[i] = (c < Ca) ? a[((long long)n * Ca + c) * hw + p]
                  : b[((long long)n * Cb + (c - Ca)) * hw + p];
}

// ----------------------------- corr lookup ---------------------------------
__global__ void k_corr_lookup(const float* __restrict__ l0, const float* __restrict__ l1,
                              const float* __restrict__ l2, const float* __restrict__ l3,
                              const float* __restrict__ coords, float* __restrict__ o)
{
  long long tot = (long long)NB * 4 * 81 * HW1;
  long long idx = blockIdx.x * 256LL + threadIdx.x;
  if (idx >= tot) return;
  int x = (int)(idx % W1); long long r = idx / W1;
  int y = (int)(r % H1);   r /= H1;
  int t = (int)(r % 81);   r /= 81;
  int lvl = (int)(r % 4);
  int b   = (int)(r / 4);
  const float* img; int hl, wl;
  switch (lvl) {
    case 0:  img = l0; hl = 60; wl = 80; break;
    case 1:  img = l1; hl = 30; wl = 40; break;
    case 2:  img = l2; hl = 15; wl = 20; break;
    default: img = l3; hl = 7;  wl = 10; break;
  }
  int n = (b * H1 + y) * W1 + x;          // pyramid image index (B*H*W, 1, hl, wl)
  img += (long long)n * hl * wl;
  float inv = 1.f / (float)(1 << lvl);
  float cx = coords[((b * 2 + 0) * H1 + y) * W1 + x];
  float cy = coords[((b * 2 + 1) * H1 + y) * W1 + x];
  int i = t / 9, j = t % 9;
  // reference quirk: delta[...,0] = DY is added to the x coordinate.
  float xs = cx * inv + (float)(i - 4);
  float ys = cy * inv + (float)(j - 4);
  float x0f = floorf(xs), y0f = floorf(ys);
  float wx1 = xs - x0f, wy1 = ys - y0f;
  float acc = 0.f;
  #pragma unroll
  for (int dy = 0; dy < 2; ++dy)
    #pragma unroll
    for (int dx = 0; dx < 2; ++dx) {
      float xi = x0f + dx, yi = y0f + dy;
      if (xi < 0.f || xi > (float)(wl - 1) || yi < 0.f || yi > (float)(hl - 1)) continue;
      float w = (dx ? wx1 : 1.f - wx1) * (dy ? wy1 : 1.f - wy1);
      acc += img[(int)yi * wl + (int)xi] * w;
    }
  o[idx] = acc;
}

// ----------------------------- convex upsample ------------------------------
__global__ void k_upsample_flow(const float* __restrict__ flow,
                                const float* __restrict__ mask,
                                float* __restrict__ out)
{
  long long tot = (long long)NB * 2 * 480 * 640;
  long long idx = blockIdx.x * 256LL + threadIdx.x;
  if (idx >= tot) return;
  int X = (int)(idx % 640); long long r = idx / 640;
  int Y = (int)(r % 480);   r /= 480;
  int ch = (int)(r % 2);
  int b  = (int)(r / 2);
  int Wq = X >> 3, xx = X & 7, Hq = Y >> 3, yy = Y & 7;
  float mv[9]; float mx = -1e30f;
  #pragma unroll
  for (int k = 0; k < 9; ++k) {
    mv[k] = mask[((long long)b * 576 + (k * 64 + yy * 8 + xx)) * HW1 + Hq * W1 + Wq];
    mx = fmaxf(mx, mv[k]);
  }
  float se = 0.f;
  #pragma unroll
  for (int k = 0; k < 9; ++k) { mv[k] = __expf(mv[k] - mx); se += mv[k]; }
  float acc = 0.f;
  #pragma unroll
  for (int k = 0; k < 9; ++k) {
    int di = k / 3 - 1, dj = k % 3 - 1;
    int yq = Hq + di, xq = Wq + dj;
    float f = 0.f;
    if (yq >= 0 && yq < H1 && xq >= 0 && xq < W1)
      f = 8.f * flow[((long long)b * 2 + ch) * HW1 + yq * W1 + xq];
    acc += mv[k] * f;
  }
  out[idx] = acc / se;
}

// ============================ host orchestration ============================

struct Bump { char* base; size_t off; };
static void* balloc(Bump& b, size_t bytes) {
  size_t a = (b.off + 255) & ~(size_t)255;
  void* p = b.base + a;
  b.off = a + bytes;
  return p;
}

struct Layer { const half_t* wh; const float* b; int Cin, Cout, kh, kw; };

static const half_t* prep_w(hipStream_t s, Bump& ws, const float* w, int Cout, int Kd) {
  int Kp = kpad(Kd);
  half_t* wh = (half_t*)balloc(ws, (size_t)Kp * Cout * sizeof(half_t));
  long long tot = (long long)Cout * Kp;
  k_wpad_f16<<<cdivu(tot, 256), 256, 0, s>>>(w, wh, Cout, Kd, Kp);
  return wh;
}

static void conv_nchw(hipStream_t s, const float* x, float* y, const Layer& L,
                      int Hin, int Win, int stride, int act, float scale,
                      half_t* colbuf)
{
  int ph = L.kh / 2, pw = L.kw / 2;
  int OH = (Hin + 2 * ph - L.kh) / stride + 1;
  int OW = (Win + 2 * pw - L.kw) / stride + 1;
  int Kd = L.Cin * L.kh * L.kw;
  int Kp = kpad(Kd);
  long long M = (long long)NB * OH * OW;
  long long tot = M * Kp;
  k_im2col_f16<<<cdivu(tot, 256), 256, 0, s>>>(x, colbuf, NB, L.Cin, Hin, Win,
                                               OH, OW, L.kh, L.kw, stride, ph, pw,
                                               Kd, Kp);
  dim3 g(cdivu(L.Cout, 64), cdivu(M, 32), 1);
  k_wmma_gemm<<<g, 256, 0, s>>>(colbuf, L.wh, L.b, y, (int)M, L.Cout, Kp,
                                0, 0, 0, scale, act, OH * OW, L.Cout);
}

static void encoder_run(hipStream_t s, const float* img, float* outp,
                        const Layer* L, half_t* colbuf, float* t1, float* t2)
{
  conv_nchw(s, img, t1, L[0], 480, 640, 2, 0, 1.f, colbuf);
  k_instnorm_relu<<<NB * 64, 256, 0, s>>>(t1, 240 * 320);
  conv_nchw(s, t1, t2, L[1], 240, 320, 2, 0, 1.f, colbuf);
  k_instnorm_relu<<<NB * 128, 256, 0, s>>>(t2, 120 * 160);
  conv_nchw(s, t2, t1, L[2], 120, 160, 2, 0, 1.f, colbuf);
  k_instnorm_relu<<<NB * 192, 256, 0, s>>>(t1, HW1);
  conv_nchw(s, t1, outp, L[3], H1, W1, 1, 0, 1.f, colbuf);
}

// d_in order (setup_inputs dict order; params in insertion order, _w then _b)
enum {
  IN_IMG1 = 0, IN_IMG2,
  P_F1W, P_F1B, P_F2W, P_F2B, P_F3W, P_F3B, P_F4W, P_F4B,
  P_C1W, P_C1B, P_C2W, P_C2B, P_C3W, P_C3B, P_C4W, P_C4B,
  P_MC1W, P_MC1B, P_MC2W, P_MC2B, P_MF1W, P_MF1B, P_MF2W, P_MF2B, P_MOW, P_MOB,
  P_HZW, P_HZB, P_HRW, P_HRB, P_HQW, P_HQB,
  P_VZW, P_VZB, P_VRW, P_VRB, P_VQW, P_VQB,
  P_FH1W, P_FH1B, P_FH2W, P_FH2B, P_MK1W, P_MK1B, P_MK2W, P_MK2B,
  IN_ITERS
};

extern "C" void kernel_launch(void* const* d_in, const int* in_sizes, int n_in,
                              void* d_out_, int out_size, void* d_ws, size_t ws_size,
                              hipStream_t stream)
{
  (void)in_sizes; (void)n_in; (void)out_size; (void)ws_size;
  hipStream_t s = stream;
  float* out = (float*)d_out_;
  const float* img1 = (const float*)d_in[IN_IMG1];
  const float* img2 = (const float*)d_in[IN_IMG2];
  #define PW(i) ((const float*)d_in[(i)])

  Bump ws{(char*)d_ws, 0};

  // ---- weights -> f16 [Cout][Kp] (B^T operand, no transpose needed) ----
  Layer Lf[4] = {
    { prep_w(s, ws, PW(P_F1W), 64, 5 * 49),    PW(P_F1B),   5,  64, 7, 7 },
    { prep_w(s, ws, PW(P_F2W), 128, 64 * 9),   PW(P_F2B),  64, 128, 3, 3 },
    { prep_w(s, ws, PW(P_F3W), 192, 128 * 9),  PW(P_F3B), 128, 192, 3, 3 },
    { prep_w(s, ws, PW(P_F4W), 256, 192),      PW(P_F4B), 192, 256, 1, 1 },
  };
  Layer Lc[4] = {
    { prep_w(s, ws, PW(P_C1W), 64, 5 * 49),    PW(P_C1B),   5,  64, 7, 7 },
    { prep_w(s, ws, PW(P_C2W), 128, 64 * 9),   PW(P_C2B),  64, 128, 3, 3 },
    { prep_w(s, ws, PW(P_C3W), 192, 128 * 9),  PW(P_C3B), 128, 192, 3, 3 },
    { prep_w(s, ws, PW(P_C4W), 256, 192),      PW(P_C4B), 192, 256, 1, 1 },
  };
  Layer Lmc1 = { prep_w(s, ws, PW(P_MC1W), 256, 324),      PW(P_MC1B), 324, 256, 1, 1 };
  Layer Lmc2 = { prep_w(s, ws, PW(P_MC2W), 192, 256 * 9),  PW(P_MC2B), 256, 192, 3, 3 };
  Layer Lmf1 = { prep_w(s, ws, PW(P_MF1W), 128, 2 * 49),   PW(P_MF1B),   2, 128, 7, 7 };
  Layer Lmf2 = { prep_w(s, ws, PW(P_MF2W), 64, 128 * 9),   PW(P_MF2B), 128,  64, 3, 3 };
  Layer Lmo  = { prep_w(s, ws, PW(P_MOW), 126, 256 * 9),   PW(P_MOB),  256, 126, 3, 3 };
  Layer Lhz  = { prep_w(s, ws, PW(P_HZW), 128, 384 * 5),   PW(P_HZB),  384, 128, 1, 5 };
  Layer Lhr  = { prep_w(s, ws, PW(P_HRW), 128, 384 * 5),   PW(P_HRB),  384, 128, 1, 5 };
  Layer Lhq  = { prep_w(s, ws, PW(P_HQW), 128, 384 * 5),   PW(P_HQB),  384, 128, 1, 5 };
  Layer Lvz  = { prep_w(s, ws, PW(P_VZW), 128, 384 * 5),   PW(P_VZB),  384, 128, 5, 1 };
  Layer Lvr  = { prep_w(s, ws, PW(P_VRW), 128, 384 * 5),   PW(P_VRB),  384, 128, 5, 1 };
  Layer Lvq  = { prep_w(s, ws, PW(P_VQW), 128, 384 * 5),   PW(P_VQB),  384, 128, 5, 1 };
  Layer Lfh1 = { prep_w(s, ws, PW(P_FH1W), 256, 128 * 9),  PW(P_FH1B), 128, 256, 3, 3 };
  Layer Lfh2 = { prep_w(s, ws, PW(P_FH2W), 2, 256 * 9),    PW(P_FH2B), 256,   2, 3, 3 };
  Layer Lmk1 = { prep_w(s, ws, PW(P_MK1W), 256, 128 * 9),  PW(P_MK1B), 128, 256, 3, 3 };
  Layer Lmk2 = { prep_w(s, ws, PW(P_MK2W), 576, 256),      PW(P_MK2B), 256, 576, 1, 1 };

  // ---- big scratch buffers ----
  half_t* colbuf = (half_t*)balloc(ws, (size_t)40000000 * sizeof(half_t));
  float*  t1     = (float*)balloc(ws, (size_t)NB * 64 * 240 * 320 * sizeof(float));
  float*  t2     = (float*)balloc(ws, (size_t)NB * 64 * 240 * 320 * sizeof(float));
  float*  fmap1  = (float*)balloc(ws, (size_t)NB * 256 * HW1 * sizeof(float));
  float*  fmap2  = (float*)balloc(ws, (size_t)NB * 256 * HW1 * sizeof(float));
  float*  cnet   = (float*)balloc(ws, (size_t)NB * 256 * HW1 * sizeof(float));
  half_t* f1T    = (half_t*)balloc(ws, (size_t)NB * HW1 * 256 * sizeof(half_t));
  half_t* f2T    = (half_t*)balloc(ws, (size_t)NB * HW1 * 256 * sizeof(half_t));
  float*  corr0  = (float*)balloc(ws, (size_t)NB * HW1 * HW1 * sizeof(float));
  float*  corr1  = (float*)balloc(ws, (size_t)NB * HW1 * 1200 * sizeof(float));
  float*  corr2  = (float*)balloc(ws, (size_t)NB * HW1 * 300 * sizeof(float));
  float*  corr3  = (float*)balloc(ws, (size_t)NB * HW1 * 70 * sizeof(float));
  float*  net    = (float*)balloc(ws, (size_t)NB * 128 * HW1 * sizeof(float));
  float*  inp    = (float*)balloc(ws, (size_t)NB * 128 * HW1 * sizeof(float));
  float*  coords1= (float*)balloc(ws, (size_t)NB * 2 * HW1 * sizeof(float));
  float*  flow   = (float*)balloc(ws, (size_t)NB * 2 * HW1 * sizeof(float));
  float*  delta  = (float*)balloc(ws, (size_t)NB * 2 * HW1 * sizeof(float));
  float*  corrf  = (float*)balloc(ws, (size_t)NB * 324 * HW1 * sizeof(float));
  float*  c1b    = (float*)balloc(ws, (size_t)NB * 256 * HW1 * sizeof(float));
  float*  c2b    = (float*)balloc(ws, (size_t)NB * 192 * HW1 * sizeof(float));
  float*  fb1    = (float*)balloc(ws, (size_t)NB * 128 * HW1 * sizeof(float));
  float*  fb2    = (float*)balloc(ws, (size_t)NB * 64 * HW1 * sizeof(float));
  float*  cf     = (float*)balloc(ws, (size_t)NB * 256 * HW1 * sizeof(float));
  float*  mbuf   = (float*)balloc(ws, (size_t)NB * 126 * HW1 * sizeof(float));
  float*  motion = (float*)balloc(ws, (size_t)NB * 128 * HW1 * sizeof(float));
  float*  xbuf   = (float*)balloc(ws, (size_t)NB * 256 * HW1 * sizeof(float));
  float*  hx     = (float*)balloc(ws, (size_t)NB * 384 * HW1 * sizeof(float));
  float*  zb     = (float*)balloc(ws, (size_t)NB * 128 * HW1 * sizeof(float));
  float*  rb     = (float*)balloc(ws, (size_t)NB * 128 * HW1 * sizeof(float));
  float*  qb     = (float*)balloc(ws, (size_t)NB * 128 * HW1 * sizeof(float));
  float*  rh     = (float*)balloc(ws, (size_t)NB * 128 * HW1 * sizeof(float));
  float*  d1     = (float*)balloc(ws, (size_t)NB * 256 * HW1 * sizeof(float));
  float*  m1     = (float*)balloc(ws, (size_t)NB * 256 * HW1 * sizeof(float));
  float*  maskb  = (float*)balloc(ws, (size_t)NB * 576 * HW1 * sizeof(float));

  // ---- encoders ----
  encoder_run(s, img1, fmap1, Lf, colbuf, t1, t2);
  encoder_run(s, img2, fmap2, Lf, colbuf, t1, t2);
  encoder_run(s, img2, cnet,  Lc, colbuf, t1, t2);   // context from image2

  // ---- correlation pyramid: corr = f1T * f2T^T (K=256), scale 1/16 ----
  for (int b = 0; b < NB; ++b) {
    k_tconv_f16<<<cdivu((long long)256 * HW1, 256), 256, 0, s>>>(
        fmap1 + (long long)b * 256 * HW1, f1T + (long long)b * HW1 * 256, 256, HW1);
    k_tconv_f16<<<cdivu((long long)256 * HW1, 256), 256, 0, s>>>(
        fmap2 + (long long)b * 256 * HW1, f2T + (long long)b * HW1 * 256, 256, HW1);
  }
  {
    dim3 g(cdivu(HW1, 64), cdivu(HW1, 32), NB);
    k_wmma_gemm<<<g, 256, 0, s>>>(f1T, f2T, nullptr, corr0, HW1, HW1, 256,
                                  (long long)HW1 * 256, (long long)HW1 * 256,
                                  (long long)HW1 * HW1, 0.0625f, 0, 0, 0);
  }
  k_avgpool2<<<cdivu((long long)NB * HW1 * 1200, 256), 256, 0, s>>>(corr0, corr1, (long long)NB * HW1, 60, 80);
  k_avgpool2<<<cdivu((long long)NB * HW1 * 300, 256), 256, 0, s>>>(corr1, corr2, (long long)NB * HW1, 30, 40);
  k_avgpool2<<<cdivu((long long)NB * HW1 * 70, 256), 256, 0, s>>>(corr2, corr3, (long long)NB * HW1, 15, 20);

  // ---- context split + coords ----
  k_split_cnet<<<cdivu((long long)NB * 128 * HW1, 256), 256, 0, s>>>(cnet, net, inp);
  k_init_coords<<<cdivu(NB * 2 * HW1, 256), 256, 0, s>>>(coords1);

  const long long nh = (long long)NB * 128 * HW1;
  const long long nc2 = (long long)NB * 2 * HW1;

  for (int it = 0; it < ITERS; ++it) {
    k_corr_lookup<<<cdivu((long long)NB * 324 * HW1, 256), 256, 0, s>>>(
        corr0, corr1, corr2, corr3, coords1, corrf);
    conv_nchw(s, corrf, c1b, Lmc1, H1, W1, 1, 1, 1.f, colbuf);
    conv_nchw(s, c1b,  c2b, Lmc2, H1, W1, 1, 1, 1.f, colbuf);
    k_sub_grid<<<cdivu(nc2, 256), 256, 0, s>>>(coords1, flow);
    conv_nchw(s, flow, fb1, Lmf1, H1, W1, 1, 1, 1.f, colbuf);
    conv_nchw(s, fb1,  fb2, Lmf2, H1, W1, 1, 1, 1.f, colbuf);
    k_concat2<<<cdivu((long long)NB * 256 * HW1, 256), 256, 0, s>>>(c2b, 192, fb2, 64, cf, HW1);
    conv_nchw(s, cf, mbuf, Lmo, H1, W1, 1, 1, 1.f, colbuf);
    k_concat2<<<cdivu((long long)NB * 128 * HW1, 256), 256, 0, s>>>(mbuf, 126, flow, 2, motion, HW1);
    k_concat2<<<cdivu((long long)NB * 256 * HW1, 256), 256, 0, s>>>(inp, 128, motion, 128, xbuf, HW1);

    // GRU horizontal (1x5)
    k_concat2<<<cdivu((long long)NB * 384 * HW1, 256), 256, 0, s>>>(net, 128, xbuf, 256, hx, HW1);
    conv_nchw(s, hx, zb, Lhz, H1, W1, 1, 3, 1.f, colbuf);
    conv_nchw(s, hx, rb, Lhr, H1, W1, 1, 3, 1.f, colbuf);
    k_ewise_mul<<<cdivu(nh, 256), 256, 0, s>>>(rb, net, rh, nh);
    k_concat2<<<cdivu((long long)NB * 384 * HW1, 256), 256, 0, s>>>(rh, 128, xbuf, 256, hx, HW1);
    conv_nchw(s, hx, qb, Lhq, H1, W1, 1, 2, 1.f, colbuf);
    k_gru_combine<<<cdivu(nh, 256), 256, 0, s>>>(net, zb, qb, nh);

    // GRU vertical (5x1)
    k_concat2<<<cdivu((long long)NB * 384 * HW1, 256), 256, 0, s>>>(net, 128, xbuf, 256, hx, HW1);
    conv_nchw(s, hx, zb, Lvz, H1, W1, 1, 3, 1.f, colbuf);
    conv_nchw(s, hx, rb, Lvr, H1, W1, 1, 3, 1.f, colbuf);
    k_ewise_mul<<<cdivu(nh, 256), 256, 0, s>>>(rb, net, rh, nh);
    k_concat2<<<cdivu((long long)NB * 384 * HW1, 256), 256, 0, s>>>(rh, 128, xbuf, 256, hx, HW1);
    conv_nchw(s, hx, qb, Lvq, H1, W1, 1, 2, 1.f, colbuf);
    k_gru_combine<<<cdivu(nh, 256), 256, 0, s>>>(net, zb, qb, nh);

    // heads
    conv_nchw(s, net, d1, Lfh1, H1, W1, 1, 1, 1.f, colbuf);
    conv_nchw(s, d1, delta, Lfh2, H1, W1, 1, 0, 1.f, colbuf);
    conv_nchw(s, net, m1, Lmk1, H1, W1, 1, 1, 1.f, colbuf);
    conv_nchw(s, m1, maskb, Lmk2, H1, W1, 1, 0, 0.25f, colbuf);

    k_ewise_add<<<cdivu(nc2, 256), 256, 0, s>>>(coords1, delta, nc2);
    k_sub_grid<<<cdivu(nc2, 256), 256, 0, s>>>(coords1, flow);
    k_upsample_flow<<<cdivu((long long)NB * 2 * 480 * 640, 256), 256, 0, s>>>(
        flow, maskb, out + 19200 + (long long)it * 1228800);
  }

  // final low-res flow -> first output
  k_sub_grid<<<cdivu(nc2, 256), 256, 0, s>>>(coords1, out);
  #undef PW
}
